// GTUBlock_37795712204994
// MI455X (gfx1250) — matrix operand order repaired and checked by reading, not compiled, not run on hardware
//
#include <hip/hip_runtime.h>
#include <hip/hip_bf16.h>
#include <math.h>

// ---------------------------------------------------------------------------
// GTU block for MI455X (gfx1250, wave32, WMMA).
//   - 6 big GEMMs via v_wmma_f32_16x16x32_bf16 (fp32 accum), fused epilogues
//   - tile staging via global_load_async_to_lds_b128 (ASYNCcnt) where possible
//   - Toeplitz matmul via per-channel 4096-pt FFT in LDS (VALU, fp32)
//   - tiny RPE MLP on VALU
// ---------------------------------------------------------------------------

typedef __attribute__((ext_vector_type(16))) __bf16 v16bf;
typedef __attribute__((ext_vector_type(8)))  float  v8f;

#define B_   4
#define N_   2048
#define EMB  512
#define H_   8
#define D1_  1536
#define HD_  192
#define M_   (B_ * N_)      // 8192 rows
#define L2N  4096           // FFT length (2*N)

#define BM 128
#define BN 64
#define BK 32

static __device__ __forceinline__ float silu_f(float x) {
    return x / (1.0f + __expf(-x));
}

// CDNA5 async copy global -> LDS (16B per lane), tracked by ASYNCcnt.
// lds = group-segment byte offset (low 32 bits of the flat shared address).
static __device__ __forceinline__ void async_copy_b128(unsigned lds,
                                                       const void* g) {
    asm volatile("global_load_async_to_lds_b128 %0, %1, off"
                 :: "v"(lds), "v"(g)
                 : "memory");
}
static __device__ __forceinline__ void wait_asynccnt0() {
    asm volatile("s_wait_asynccnt 0x0" ::: "memory");
}
static __device__ __forceinline__ unsigned lds_off(const void* p) {
    return (unsigned)(uintptr_t)p;   // flat->LDS addrspace cast == truncate
}

// ---------------------------------------------------------------------------
// 1) SimpleRMSNorm over EMB=512 features, divisor uses D1=1536 (faithful to
//    the reference bug), output cast to bf16 for WMMA.
// ---------------------------------------------------------------------------
__global__ void __launch_bounds__(256)
rmsnorm_cast_kernel(const float* __restrict__ x, __bf16* __restrict__ xnb) {
    __shared__ float red[256];
    const int row = blockIdx.x, tid = threadIdx.x;
    const float* xp = x + (size_t)row * EMB;
    float a = xp[tid], b = xp[tid + 256];
    red[tid] = a * a + b * b;
    __syncthreads();
    for (int s = 128; s > 0; s >>= 1) {
        if (tid < s) red[tid] += red[tid + s];
        __syncthreads();
    }
    // 1536^-0.5 = 0.02551551815...
    const float scale = 1.0f / (sqrtf(red[0]) * 0.02551551815399144f + 1e-8f);
    __bf16* o = xnb + (size_t)row * EMB;
    o[tid]       = (__bf16)(a * scale);
    o[tid + 256] = (__bf16)(b * scale);
}

// ---------------------------------------------------------------------------
// 2) fp32 [K,N] weight -> bf16 transposed [N,K] (makes WMMA B-fragments
//    pairs of 16B-contiguous runs in global memory).
// ---------------------------------------------------------------------------
__global__ void castT_kernel(const float* __restrict__ w, __bf16* __restrict__ wt,
                             int K, int N) {
    int i = blockIdx.x * 256 + threadIdx.x;
    if (i < K * N) {
        int k = i / N, n = i - k * N;
        wt[(size_t)n * K + k] = (__bf16)w[i];
    }
}

// ---------------------------------------------------------------------------
// 3) RPE MLP: one block per position k in [0,4096). a[h][k][d] fp32.
//    k=0,2048 -> idx 0; k in [1,2048) -> idx k; else idx k-4096.
// ---------------------------------------------------------------------------
__global__ void __launch_bounds__(256)
rpe_kernel(const float* __restrict__ pw, const float* __restrict__ pb,
           const float* __restrict__ lw, const float* __restrict__ lb,
           const float* __restrict__ ow, const float* __restrict__ ob,
           float* __restrict__ a) {
    __shared__ float hbuf[32], rbuf[32];
    __shared__ float sscale;
    const int k = blockIdx.x, tid = threadIdx.x;
    float idx;
    if (k == 0 || k == 2048)      idx = 0.0f;
    else if (k < 2048)            idx = (float)k;
    else                          idx = (float)(k - 4096);

    if (tid < 32) hbuf[tid] = fmaxf(idx * pw[tid] + pb[tid], 0.0f);
    __syncthreads();

    for (int layer = 0; layer < 4; ++layer) {
        if (tid == 0) {
            float s = 0.0f;
            for (int j = 0; j < 32; ++j) s += hbuf[j] * hbuf[j];
            // 32^-0.5 = 0.17677669529...
            sscale = 1.0f / (sqrtf(s) * 0.1767766952966369f + 1e-8f);
        }
        __syncthreads();
        if (tid < 32) rbuf[tid] = fmaxf(hbuf[tid] * sscale, 0.0f);
        __syncthreads();
        if (layer < 3) {
            if (tid < 32) {
                float s = lb[layer * 32 + tid];
                for (int j = 0; j < 32; ++j)
                    s += rbuf[j] * lw[(layer * 32 + j) * 32 + tid];
                hbuf[tid] = s;
            }
            __syncthreads();
        }
    }
    // final 32 -> 1536 projection; scatter to a[h][k][d]
    for (int o = tid; o < D1_; o += 256) {
        float s = ob[o];
        for (int j = 0; j < 32; ++j) s += rbuf[j] * ow[j * D1_ + o];
        int hh = o / HD_, d = o - hh * HD_;
        a[((size_t)hh * L2N + k) * HD_ + d] = s;
    }
}

// ---------------------------------------------------------------------------
// 4) In-LDS 4096-pt complex FFT (iterative DIT, bit-reversal). dir=-1 fwd,
//    +1 inv (caller scales by 1/4096). 256 threads, fp32, 32 KB LDS.
// ---------------------------------------------------------------------------
__device__ __forceinline__ void block_fft4096(float2* S, int tid, int dir) {
    for (int i = tid; i < L2N; i += 256) {
        unsigned j = __brev((unsigned)i) >> 20;   // 12-bit reversal
        if ((int)j > i) { float2 t = S[i]; S[i] = S[j]; S[j] = t; }
    }
    __syncthreads();
    for (int len = 2; len <= L2N; len <<= 1) {
        const int halfl = len >> 1;
        const float ang = (float)dir * 6.283185307179586f / (float)len;
        for (int bfy = tid; bfy < (L2N / 2); bfy += 256) {
            int grp = bfy / halfl;
            int pos = bfy - grp * halfl;
            int i0 = grp * len + pos;
            int i1 = i0 + halfl;
            float sn, cs;
            __sincosf(ang * (float)pos, &sn, &cs);
            float2 u = S[i0], w = S[i1];
            float tr = w.x * cs - w.y * sn;
            float ti = w.x * sn + w.y * cs;
            S[i0] = make_float2(u.x + tr, u.y + ti);
            S[i1] = make_float2(u.x - tr, u.y - ti);
        }
        __syncthreads();
    }
}

// FFT of the Toeplitz kernels: one block per channel c = h*HD+d.
__global__ void __launch_bounds__(256)
fftA_kernel(const float* __restrict__ a, float2* __restrict__ AF) {
    __shared__ float2 S[L2N];
    const int bid = blockIdx.x;          // c = h*HD + d
    const int d = bid % HD_, h = bid / HD_;
    const int tid = threadIdx.x;
    const float* ap = a + (size_t)h * L2N * HD_ + d;
    for (int k = tid; k < L2N; k += 256)
        S[k] = make_float2(ap[(size_t)k * HD_], 0.0f);
    __syncthreads();
    block_fft4096(S, tid, -1);
    float2* out = AF + (size_t)bid * L2N;
    for (int k = tid; k < L2N; k += 256) out[k] = S[k];
}

// Circular conv: t[b,:,c] = irfft(rfft(pad(v[b,:,c])) * AF[c])[0:N].
// One block per (b, c); v/t stored [M, D1] row-major. The stride-D1 column
// reads/writes are L2-friendly: 16 neighbouring channel-blocks share each
// 64B line and v_f (50MB) fits in the 192MB L2.
__global__ void __launch_bounds__(256)
conv_kernel(const float* __restrict__ v, const float2* __restrict__ AF,
            float* __restrict__ t) {
    __shared__ float2 S[L2N];
    const int bid = blockIdx.x;          // b*D1 + c
    const int c = bid % D1_;
    const int b = bid / D1_;
    const int tid = threadIdx.x;
    const float* vp = v + (size_t)b * N_ * D1_ + c;
    for (int n = tid; n < L2N; n += 256)
        S[n] = (n < N_) ? make_float2(vp[(size_t)n * D1_], 0.0f)
                        : make_float2(0.0f, 0.0f);
    __syncthreads();
    block_fft4096(S, tid, -1);
    const float2* af = AF + (size_t)c * L2N;
    for (int k = tid; k < L2N; k += 256) {
        float2 s = S[k], aa = af[k];
        S[k] = make_float2(s.x * aa.x - s.y * aa.y, s.x * aa.y + s.y * aa.x);
    }
    __syncthreads();
    block_fft4096(S, tid, +1);
    float* tp = t + (size_t)b * N_ * D1_ + c;
    const float inv = 1.0f / (float)L2N;
    for (int n = tid; n < N_; n += 256) tp[(size_t)n * D1_] = S[n].x * inv;
}

// ---------------------------------------------------------------------------
// 5) WMMA GEMM: C[M,N] = A[M,K] @ W[K,N] (+ fused epilogue). bf16 in, fp32
//    accum. 256 thr = 8 waves, 128x64 tile, K-step 32. Tiles staged into LDS
//    in WMMA fragment order (ISA 7.12.2) so fragment loads are ds_load_b128;
//    staging uses global_load_async_to_lds_b128 (no VGPR round trip).
//    MODE 0: A=xn(bf16), dual W (u_w,v_w)  -> u=silu(.+b), v=silu(.+b)  fp32
//    MODE 1: A=u*t (fp32*fp32 -> bf16), W=o_w -> out=.+b (fp32 to d_out),
//            x2b = bf16(out + x)
//    MODE 2: A=x2b, dual W (glu1,glu2)     -> gp = bf16(silu(.+b1)*(.+b2))
//    MODE 3: A=gp,  W=glu3                 -> g = .+b (fp32 to d_out)
// ---------------------------------------------------------------------------
template <int MODE>
__global__ void __launch_bounds__(256)
gemm_bf16(const __bf16* __restrict__ Abf,
          const float* __restrict__ Af0, const float* __restrict__ Af1,
          const __bf16* __restrict__ W0T, const __bf16* __restrict__ W1T,
          const float* __restrict__ bias0, const float* __restrict__ bias1,
          float* __restrict__ O0, float* __restrict__ O1,
          __bf16* __restrict__ Obf, const float* __restrict__ resid,
          int K, int N) {
    constexpr bool DUAL = (MODE == 0 || MODE == 2);

    __shared__ alignas(32) __bf16 As [8][32 * 16];   // 8 KB, fragment order
    __shared__ alignas(32) __bf16 Bs0[4][32 * 16];   // 4 KB
    __shared__ alignas(32) __bf16 Bs1[4][32 * 16];   // 4 KB

    const int tid  = threadIdx.x;
    const int lane = tid & 31;
    const int wid  = tid >> 5;
    const int wm   = wid & 3;     // 4 wave-rows of 32
    const int wn   = wid >> 2;    // 2 wave-cols of 32
    const int half = lane >> 4;
    const int lr   = lane & 15;

    const int m0 = blockIdx.y * BM;
    const int n0 = blockIdx.x * BN;

    // A loader: thread -> (subtile 0..7, fragment lane 0..31)
    const int lsub  = tid >> 5;
    const int ll    = tid & 31;
    const int arow  = m0 + lsub * 16 + (ll & 15);
    const int ahalf = ll >> 4;

    // B loader: 128 (subtile, lane) pairs per weight
    const int bsub  = (tid & 127) >> 5;
    const int bl    = tid & 31;
    const int bcol  = n0 + bsub * 16 + (bl & 15);
    const int bhalf = bl >> 4;
    const bool bsel1 = (tid >= 128);

    // per-lane LDS destinations for the async copies
    const unsigned aLds = lds_off(&As[lsub][ll * 16]);
    const unsigned bLds = lds_off((DUAL && bsel1) ? &Bs1[bsub][bl * 16]
                                                  : &Bs0[bsub][bl * 16]);

    v8f acc0[2][2] = {};
    v8f acc1[2][2] = {};

    for (int kb = 0; kb < K; kb += BK) {
        // ---- stage A tile (fragment order: e<8 -> K=kb+8h+e ; e>=8 -> +16)
        if constexpr (MODE == 1) {
            const size_t base = (size_t)arow * K + kb + 8 * ahalf;
            const float* up = Af0 + base;
            const float* tp = Af1 + base;
            __bf16* dst = &As[lsub][ll * 16];
#pragma unroll
            for (int e = 0; e < 8; ++e) dst[e]     = (__bf16)(up[e]      * tp[e]);
#pragma unroll
            for (int e = 0; e < 8; ++e) dst[8 + e] = (__bf16)(up[16 + e] * tp[16 + e]);
        } else {
            const __bf16* ap = Abf + (size_t)arow * K + kb + 8 * ahalf;
            async_copy_b128(aLds,      ap);        // 8 bf16, e=0..7
            async_copy_b128(aLds + 16, ap + 16);   // 8 bf16, e=8..15
            if (kb + BK < K) __builtin_prefetch(ap + BK, 0, 0);
        }
        // ---- stage B tile(s): W^T[n][k] -> fragment order (e<8 -> K=kb+8h+e ; e>=8 -> +16)
        if (DUAL || tid < 128) {
            const __bf16* wsrc = (DUAL && bsel1) ? W1T : W0T;
            const __bf16* wp = wsrc + (size_t)bcol * K + kb + 8 * bhalf;
            async_copy_b128(bLds,      wp);
            async_copy_b128(bLds + 16, wp + 16);
        }
        wait_asynccnt0();
        __syncthreads();

        v16bf afr[2], b0fr[2], b1fr[2];
#pragma unroll
        for (int t = 0; t < 2; ++t) {
            afr[t]  = *(const v16bf*)&As [wm * 2 + t][lane * 16];
            b0fr[t] = *(const v16bf*)&Bs0[wn * 2 + t][lane * 16];
            if constexpr (DUAL)
                b1fr[t] = *(const v16bf*)&Bs1[wn * 2 + t][lane * 16];
        }
#pragma unroll
        for (int tm = 0; tm < 2; ++tm)
#pragma unroll
            for (int tn = 0; tn < 2; ++tn) {
                acc0[tm][tn] = __builtin_amdgcn_wmma_f32_16x16x32_bf16(
                    false, afr[tm], false, b0fr[tn], (short)0, acc0[tm][tn],
                    false, false);
                if constexpr (DUAL)
                    acc1[tm][tn] = __builtin_amdgcn_wmma_f32_16x16x32_bf16(
                        false, afr[tm], false, b1fr[tn], (short)0, acc1[tm][tn],
                        false, false);
            }
        __syncthreads();
    }

    // ---- epilogue: C layout — VGPR p holds row p (lanes 0-15) / p+8 (16-31)
#pragma unroll
    for (int tm = 0; tm < 2; ++tm)
#pragma unroll
        for (int tn = 0; tn < 2; ++tn)
#pragma unroll
            for (int p = 0; p < 8; ++p) {
                const int row = m0 + wm * 32 + tm * 16 + p + 8 * half;
                const int col = n0 + wn * 32 + tn * 16 + lr;
                const size_t idx = (size_t)row * N + col;
                const float v0 = acc0[tm][tn][p];
                if constexpr (MODE == 0) {
                    O0[idx] = silu_f(v0 + bias0[col]);
                    O1[idx] = silu_f(acc1[tm][tn][p] + bias1[col]);
                } else if constexpr (MODE == 1) {
                    const float o = v0 + bias0[col];
                    O0[idx]  = o;                                   // "out"
                    Obf[idx] = (__bf16)(o + resid[idx]);            // x2
                } else if constexpr (MODE == 2) {
                    const float g1 = silu_f(v0 + bias0[col]);
                    const float g2 = acc1[tm][tn][p] + bias1[col];
                    Obf[idx] = (__bf16)(g1 * g2);
                } else {
                    O0[idx] = v0 + bias0[col];                      // "g"
                }
            }
}

// ---------------------------------------------------------------------------
// Host orchestration
// ---------------------------------------------------------------------------
extern "C" void kernel_launch(void* const* d_in, const int* in_sizes, int n_in,
                              void* d_out, int out_size, void* d_ws,
                              size_t ws_size, hipStream_t stream) {
    (void)in_sizes; (void)n_in; (void)out_size; (void)ws_size;

    const float* x   = (const float*)d_in[1];     // d_in[0] = xyz (unused)
    const float* u_w = (const float*)d_in[2];
    const float* u_b = (const float*)d_in[3];
    const float* v_w = (const float*)d_in[4];
    const float* v_b = (const float*)d_in[5];
    const float* o_w = (const float*)d_in[6];
    const float* o_b = (const float*)d_in[7];
    const float* rpw = (const float*)d_in[8];
    const float* rpb = (const float*)d_in[9];
    const float* rlw = (const float*)d_in[10];
    const float* rlb = (const float*)d_in[11];
    const float* row = (const float*)d_in[12];
    const float* rob = (const float*)d_in[13];
    const float* g1w = (const float*)d_in[14];
    const float* g1b = (const float*)d_in[15];
    const float* g2w = (const float*)d_in[16];
    const float* g2b = (const float*)d_in[17];
    const float* g3w = (const float*)d_in[18];
    const float* g3b = (const float*)d_in[19];

    char* ws = (char*)d_ws;
    size_t off = 0;
    auto take = [&](size_t bytes) -> char* {
        char* p = ws + off;
        off = (off + bytes + 255) & ~(size_t)255;
        return p;
    };

    __bf16* xnb = (__bf16*)take((size_t)M_ * EMB * 2);
    __bf16* uwT = (__bf16*)take((size_t)D1_ * EMB * 2);
    __bf16* vwT = (__bf16*)take((size_t)D1_ * EMB * 2);
    __bf16* owT = (__bf16*)take((size_t)EMB * D1_ * 2);
    __bf16* g1T = (__bf16*)take((size_t)D1_ * EMB * 2);
    __bf16* g2T = (__bf16*)take((size_t)D1_ * EMB * 2);
    __bf16* g3T = (__bf16*)take((size_t)EMB * D1_ * 2);
    float*  u_f = (float*)take((size_t)M_ * D1_ * 4);
    float*  v_f = (float*)take((size_t)M_ * D1_ * 4);
    float*  t_f = (float*)take((size_t)M_ * D1_ * 4);
    float*  a_f = (float*)take((size_t)H_ * L2N * HD_ * 4);
    float2* AF  = (float2*)take((size_t)D1_ * L2N * 8);
    __bf16* x2b = (__bf16*)take((size_t)M_ * EMB * 2);
    __bf16* gpb = (__bf16*)take((size_t)M_ * D1_ * 2);

    const int nwt = (EMB * D1_ + 255) / 256;
    castT_kernel<<<nwt, 256, 0, stream>>>(u_w, uwT, EMB, D1_);
    castT_kernel<<<nwt, 256, 0, stream>>>(v_w, vwT, EMB, D1_);
    castT_kernel<<<nwt, 256, 0, stream>>>(o_w, owT, D1_, EMB);
    castT_kernel<<<nwt, 256, 0, stream>>>(g1w, g1T, EMB, D1_);
    castT_kernel<<<nwt, 256, 0, stream>>>(g2w, g2T, EMB, D1_);
    castT_kernel<<<nwt, 256, 0, stream>>>(g3w, g3T, D1_, EMB);

    rmsnorm_cast_kernel<<<M_, 256, 0, stream>>>(x, xnb);
    rpe_kernel<<<L2N, 256, 0, stream>>>(rpw, rpb, rlw, rlb, row, rob, a_f);
    fftA_kernel<<<D1_, 256, 0, stream>>>(a_f, AF);

    dim3 gBig(D1_ / BN, M_ / BM);   // 24 x 64
    dim3 gSml(EMB / BN, M_ / BM);   //  8 x 64

    // u = silu(xn@u_w+b), v = silu(xn@v_w+b)
    gemm_bf16<0><<<gBig, 256, 0, stream>>>(xnb, nullptr, nullptr, uwT, vwT,
                                           u_b, v_b, u_f, v_f, nullptr,
                                           nullptr, EMB, D1_);
    // Toeplitz via FFT: t = irfft(rfft(v,2n)*rfft(a,2n))[:n]
    conv_kernel<<<B_ * D1_, 256, 0, stream>>>(v_f, AF, t_f);

    float* g_out = (float*)d_out;                 // output 0: g  [M, EMB]
    float* o_out = g_out + (size_t)M_ * EMB;      // output 1: out [M, EMB]

    // out = (u*t)@o_w + o_b ; x2 = out + x (bf16)
    gemm_bf16<1><<<gSml, 256, 0, stream>>>(nullptr, u_f, t_f, owT, nullptr,
                                           o_b, nullptr, o_out, nullptr, x2b,
                                           x, D1_, EMB);
    // gp = silu(x2@g1+b1) * (x2@g2+b2)
    gemm_bf16<2><<<gBig, 256, 0, stream>>>(x2b, nullptr, nullptr, g1T, g2T,
                                           g1b, g2b, nullptr, nullptr, gpb,
                                           nullptr, EMB, D1_);
    // g = gp@g3 + b3
    gemm_bf16<3><<<gSml, 256, 0, stream>>>(gpb, nullptr, nullptr, g3T, nullptr,
                                           g3b, nullptr, g_out, nullptr,
                                           nullptr, nullptr, D1_, EMB);
}
